// GNNJointInterpNetwork_21174188770102
// MI455X (gfx1250) — matrix operand order repaired and checked by reading, not compiled, not run on hardware
//
#include <hip/hip_runtime.h>
#include <hip/hip_bf16.h>

// ---------------------------------------------------------------------------
// Types for WMMA bf16 (gfx1250: v_wmma_f32_16x16x32_bf16, wave32)
// ---------------------------------------------------------------------------
typedef __attribute__((ext_vector_type(16))) __bf16 v16bf;
typedef __attribute__((ext_vector_type(8)))  float  v8f;

struct U128 { unsigned x, y, z, w; };
union Frag { v16bf v; U128 q[2]; };

__device__ __forceinline__ unsigned short f2bf(float f) {
  unsigned u = __float_as_uint(f);
  u += 0x7fffu + ((u >> 16) & 1u);   // round-to-nearest-even
  return (unsigned short)(u >> 16);
}

// ---------------------------------------------------------------------------
// Generic GEMM:  C[Mr,Nc] = act( A(bf16, Mr x K, lda) @ BT(bf16, Nc x K, ldb)^T + bias )
// One wave (32 threads) computes one 16x16 tile via v_wmma_f32_16x16x32_bf16.
// Optionally writes f32 (ldcf) and/or bf16 (ldcb) outputs.
// act: 0 = none, 1 = relu, 2 = sigmoid
// ---------------------------------------------------------------------------
__global__ void gemm_wmma(const unsigned short* __restrict__ A, int lda,
                          const unsigned short* __restrict__ BT, int ldb,
                          const float* __restrict__ bias,
                          float* __restrict__ Cf, int ldcf,
                          unsigned short* __restrict__ Cb, int ldcb,
                          int K, int act) {
  const int mt   = blockIdx.x;
  const int nt   = blockIdx.y;
  const int lane = threadIdx.x;      // 0..31
  const int half = lane >> 4;        // 0/1
  const int l16  = lane & 15;

  const int arow = mt * 16 + l16;    // A fragment: lane holds row m = lane%16
  const int bcol = nt * 16 + l16;    // B fragment: lane holds col n = lane%16

  v8f acc;
  const float bv = bias ? bias[bcol] : 0.0f;
#pragma unroll
  for (int r = 0; r < 8; ++r) acc[r] = bv;

  const unsigned short* ap = A  + (size_t)arow * lda;
  const unsigned short* bp = BT + (size_t)bcol * ldb;

  for (int kb = 0; kb < K; kb += 32) {
    Frag a, b;
    // A 16x32 bf16 layout (ISA 7.12.2): v[0..7] -> K = half*8 + 0..7,
    //                                   v[8..15] -> K = 16 + half*8 + 0..7
    const int ao = kb + half * 8;
    a.q[0] = *(const U128*)(ap + ao);
    a.q[1] = *(const U128*)(ap + ao + 16);
    // B 32x16 bf16 layout (same K order as A): v[0..7] -> K = half*8 + 0..7,
    //                                          v[8..15] -> K = 16 + half*8 + 0..7
    const int bo = kb + half * 8;
    b.q[0] = *(const U128*)(bp + bo);
    b.q[1] = *(const U128*)(bp + bo + 16);
    acc = __builtin_amdgcn_wmma_f32_16x16x32_bf16(false, a.v, false, b.v,
                                                  (short)0, acc, false, false);
  }

  const int orow = mt * 16 + half * 8;   // C: VGPR r -> M = 8*(lane/16)+r, N = lane%16
#pragma unroll
  for (int r = 0; r < 8; ++r) {
    float v = acc[r];
    if (act == 1) v = v > 0.0f ? v : 0.0f;
    else if (act == 2) v = 1.0f / (1.0f + __expf(-v));
    const int rr = orow + r;
    if (Cf) Cf[(size_t)rr * ldcf + bcol] = v;
    if (Cb) Cb[(size_t)rr * ldcb + bcol] = f2bf(v);
  }
}

// ---------------------------------------------------------------------------
// Prep kernels
// ---------------------------------------------------------------------------
// out[n*K + k] = bf16( in[(row0+k)*N + n] )   (transpose + convert)
__global__ void w_t_bf16(const float* __restrict__ in, unsigned short* __restrict__ out,
                         int K, int N, int row0) {
  int idx = blockIdx.x * blockDim.x + threadIdx.x;
  if (idx >= N * K) return;
  int n = idx / K, k = idx % K;
  out[idx] = f2bf(in[(size_t)(row0 + k) * N + n]);
}

// x:(T,H,D) f32 -> xc[t,d,h]=x[t,h,d], xn[t,d,h]=x[t+1,h,d]   (Tm=16,D=128,H=256)
__global__ void x_t_bf16(const float* __restrict__ x,
                         unsigned short* __restrict__ xc,
                         unsigned short* __restrict__ xn) {
  int idx = blockIdx.x * blockDim.x + threadIdx.x;   // Tm*D*H = 524288
  if (idx >= 16 * 128 * 256) return;
  int t = idx / (128 * 256);
  int r = idx % (128 * 256);
  int d = r / 256, h = r % 256;
  xc[idx] = f2bf(x[((size_t)t       * 256 + h) * 128 + d]);
  xn[idx] = f2bf(x[((size_t)(t + 1) * 256 + h) * 128 + d]);
}

// e_bf[ij,e] = bf16( relu(edge_w[ij]*We[e] + be[e]) )
__global__ void edge_enc(const float* __restrict__ edge_w, const float* __restrict__ We,
                         const float* __restrict__ be, unsigned short* __restrict__ ebf) {
  int idx = blockIdx.x * blockDim.x + threadIdx.x;   // 16384*32
  if (idx >= 16384 * 32) return;
  int ij = idx >> 5, e = idx & 31;
  float v = edge_w[ij] * We[e] + be[e];
  ebf[idx] = f2bf(v > 0.0f ? v : 0.0f);
}

// ---------------------------------------------------------------------------
// Masked max aggregation:
// agg[t,i,m] = max_{j: edge_w[i,j]>0 || j==i} relu(A[t,j,m] + B[i,j,m])
// (A already includes bm bias; relu>=0 makes acc=0 init exact)
// Writes bf16 into cat[:,128:256].
// ---------------------------------------------------------------------------
__global__ void msg_maxagg(const float* __restrict__ Amsg,   // (Tm*128,128)
                           const float* __restrict__ Bmsg,   // (16384,128)
                           const float* __restrict__ edge_w, // (128,128)
                           unsigned short* __restrict__ cat) { // (Tm*128,256)
  const int i = blockIdx.x, t = blockIdx.y, m = threadIdx.x;   // 128 threads
  const float* At = Amsg + (size_t)t * 128 * 128 + m;
  const float* Bi = Bmsg + (size_t)i * 128 * 128 + m;
  float acc = 0.0f;
  for (int j = 0; j < 128; ++j) {
    float w = edge_w[i * 128 + j];
    if (w > 0.0f || j == i) {
      float v = At[(size_t)j * 128] + Bi[(size_t)j * 128];
      v = v > 0.0f ? v : 0.0f;
      acc = fmaxf(acc, v);
    }
  }
  cat[((size_t)t * 128 + i) * 256 + 128 + m] = f2bf(acc);
}

// h' = g*upd + (1-g)*h  -> bf16 into df[:, coloff : coloff+128]
__global__ void gate_combine(const float* __restrict__ g, const float* __restrict__ upd,
                             const float* __restrict__ h,
                             unsigned short* __restrict__ df, int coloff) {
  int idx = blockIdx.x * blockDim.x + threadIdx.x;   // 2048*128
  if (idx >= 2048 * 128) return;
  int row = idx >> 7, m = idx & 127;
  float gv = g[idx];
  float v = gv * upd[idx] + (1.0f - gv) * h[idx];
  df[(size_t)row * 256 + coloff + m] = f2bf(v);
}

// logits[t,i,j] = bc2 + sum_m relu(Na[t,i,m]+Cb[t,j,m]+Eb[i,j,m]) * Wc2[m]
// (Eb already includes bc1); bad-edge mask -> -1e9
__global__ void logits_kernel(const float* __restrict__ Na, const float* __restrict__ Cb,
                              const float* __restrict__ Eb, const float* __restrict__ Wc2,
                              const float* __restrict__ bc2, const float* __restrict__ edge_w,
                              float* __restrict__ out) {
  __shared__ float sNa[128], sW[128];
  const int i = blockIdx.x, t = blockIdx.y, j = threadIdx.x;   // 128 threads
  sNa[j] = Na[((size_t)t * 128 + i) * 128 + j];
  sW[j]  = Wc2[j];
  __syncthreads();
  const float* cb = Cb + ((size_t)t * 128 + j) * 128;
  const float* eb = Eb + ((size_t)i * 128 + j) * 128;
  float acc = bc2[0];
  for (int m = 0; m < 128; m += 4) {
    float4 c = *(const float4*)(cb + m);
    float4 e = *(const float4*)(eb + m);
    float v;
    v = sNa[m + 0] + c.x + e.x; acc += fmaxf(v, 0.0f) * sW[m + 0];
    v = sNa[m + 1] + c.y + e.y; acc += fmaxf(v, 0.0f) * sW[m + 1];
    v = sNa[m + 2] + c.z + e.z; acc += fmaxf(v, 0.0f) * sW[m + 2];
    v = sNa[m + 3] + c.w + e.w; acc += fmaxf(v, 0.0f) * sW[m + 3];
  }
  float w = edge_w[i * 128 + j];
  if (w == 0.0f && i != j) acc = -1e9f;
  out[((size_t)t * 128 + i) * 128 + j] = acc;
}

// dist[row] = dot(dh[row,:128], Wd2) + bd2   (one wave32 per row, 4 rows/block)
__global__ void dist_kernel(const float* __restrict__ dh, const float* __restrict__ Wd2,
                            const float* __restrict__ bd2, float* __restrict__ out) {
  const int row  = blockIdx.x * 4 + (threadIdx.x >> 5);
  const int lane = threadIdx.x & 31;
  const float4 v = *(const float4*)(dh + (size_t)row * 128 + lane * 4);
  const float4 w = *(const float4*)(Wd2 + lane * 4);
  float s = v.x * w.x + v.y * w.y + v.z * w.z + v.w * w.w;
  for (int off = 16; off > 0; off >>= 1) s += __shfl_xor(s, off, 32);
  if (lane == 0) out[row] = s + bd2[0];
}

// ---------------------------------------------------------------------------
// Host launcher
// ---------------------------------------------------------------------------
extern "C" void kernel_launch(void* const* d_in, const int* in_sizes, int n_in,
                              void* d_out, int out_size, void* d_ws, size_t ws_size,
                              hipStream_t stream) {
  (void)in_sizes; (void)n_in; (void)out_size; (void)ws_size;
  const float* x      = (const float*)d_in[0];
  const float* edge_w = (const float*)d_in[1];
  const float* Wn  = (const float*)d_in[5];  const float* bn  = (const float*)d_in[6];
  const float* We  = (const float*)d_in[7];  const float* be  = (const float*)d_in[8];
  const float* Wm  = (const float*)d_in[9];  const float* bm  = (const float*)d_in[10];
  const float* Wu  = (const float*)d_in[11]; const float* bu  = (const float*)d_in[12];
  const float* Wg  = (const float*)d_in[13]; const float* bg  = (const float*)d_in[14];
  const float* Wc1 = (const float*)d_in[15]; const float* bc1 = (const float*)d_in[16];
  const float* Wc2 = (const float*)d_in[17]; const float* bc2 = (const float*)d_in[18];
  const float* Wd1 = (const float*)d_in[19]; const float* bd1 = (const float*)d_in[20];
  const float* Wd2 = (const float*)d_in[21]; const float* bd2 = (const float*)d_in[22];

  char* base = (char*)d_ws; size_t off = 0;
  auto alloc = [&](size_t bytes) -> void* {
    void* p = base + off; off = (off + bytes + 255) & ~(size_t)255; return p;
  };

  // bf16 buffers (ushort)
  unsigned short* wn_t   = (unsigned short*)alloc(256 * 128 * 2);
  unsigned short* wmh_t  = (unsigned short*)alloc(128 * 128 * 2);
  unsigned short* wme_t  = (unsigned short*)alloc(128 * 32  * 2);
  unsigned short* wu_t   = (unsigned short*)alloc(128 * 256 * 2);
  unsigned short* wg_t   = (unsigned short*)alloc(128 * 256 * 2);
  unsigned short* wc1a_t = (unsigned short*)alloc(128 * 128 * 2);
  unsigned short* wc1b_t = (unsigned short*)alloc(128 * 128 * 2);
  unsigned short* wc1c_t = (unsigned short*)alloc(128 * 32  * 2);
  unsigned short* wd1_t  = (unsigned short*)alloc(128 * 256 * 2);
  unsigned short* xc_bf  = (unsigned short*)alloc(2048 * 256 * 2);
  unsigned short* xn_bf  = (unsigned short*)alloc(2048 * 256 * 2);
  unsigned short* e_bf   = (unsigned short*)alloc(16384 * 32 * 2);
  unsigned short* cat_c  = (unsigned short*)alloc(2048 * 256 * 2);
  unsigned short* cat_n  = (unsigned short*)alloc(2048 * 256 * 2);
  unsigned short* dfbuf  = (unsigned short*)alloc(2048 * 256 * 2);
  // f32 buffers
  float* h0_c  = (float*)alloc(2048 * 128 * 4);
  float* h0_n  = (float*)alloc(2048 * 128 * 4);
  float* A_c   = (float*)alloc(2048 * 128 * 4);
  float* A_n   = (float*)alloc(2048 * 128 * 4);
  float* upd_c = (float*)alloc(2048 * 128 * 4);
  float* g_c   = (float*)alloc(2048 * 128 * 4);
  float* upd_n = (float*)alloc(2048 * 128 * 4);
  float* g_n   = (float*)alloc(2048 * 128 * 4);
  float* Bmsg  = (float*)alloc(16384 * 128 * 4);
  // reuse (consumed before reuse point)
  float* Eb = Bmsg; float* Na = upd_c; float* Cb = g_c; float* dh = upd_n;

  auto tcv = [&](const float* in, unsigned short* out, int K, int N, int row0) {
    int n = K * N;
    w_t_bf16<<<dim3((n + 255) / 256), dim3(256), 0, stream>>>(in, out, K, N, row0);
  };
  auto gemm = [&](const unsigned short* A, int lda, const unsigned short* BT, int ldb,
                  const float* bias, float* Cf, int ldcf, unsigned short* Cbf, int ldcb,
                  int Mr, int Nc, int K, int act) {
    gemm_wmma<<<dim3(Mr / 16, Nc / 16), dim3(32), 0, stream>>>(
        A, lda, BT, ldb, bias, Cf, ldcf, Cbf, ldcb, K, act);
  };

  // --- prep: weights (transposed bf16), x (transposed bf16), edge encoder ---
  tcv(Wn,  wn_t,   256, 128, 0);
  tcv(Wm,  wmh_t,  128, 128, 0);
  tcv(Wm,  wme_t,  32,  128, 128);
  tcv(Wu,  wu_t,   256, 128, 0);
  tcv(Wg,  wg_t,   256, 128, 0);
  tcv(Wc1, wc1a_t, 128, 128, 0);
  tcv(Wc1, wc1b_t, 128, 128, 128);
  tcv(Wc1, wc1c_t, 32,  128, 256);
  tcv(Wd1, wd1_t,  256, 128, 0);
  x_t_bf16<<<dim3((16 * 128 * 256 + 255) / 256), dim3(256), 0, stream>>>(x, xc_bf, xn_bf);
  edge_enc<<<dim3((16384 * 32 + 255) / 256), dim3(256), 0, stream>>>(edge_w, We, be, e_bf);

  // --- shared: B_msg = e_enc @ Wm_e  (16384 x 32 x 128) ---
  gemm(e_bf, 32, wme_t, 32, nullptr, Bmsg, 128, nullptr, 0, 16384, 128, 32, 0);

  // --- per-stream GNN ---
  // stream "curr"
  gemm(xc_bf, 256, wn_t, 256, bn, h0_c, 128, cat_c, 256, 2048, 128, 256, 1);  // h = relu(x@Wn+bn)
  gemm(cat_c, 256, wmh_t, 128, bm, A_c, 128, nullptr, 0, 2048, 128, 128, 0);  // A = h@Wm_h + bm
  msg_maxagg<<<dim3(128, 16), dim3(128), 0, stream>>>(A_c, Bmsg, edge_w, cat_c);
  gemm(cat_c, 256, wu_t, 256, bu, upd_c, 128, nullptr, 0, 2048, 128, 256, 1); // relu
  gemm(cat_c, 256, wg_t, 256, bg, g_c, 128, nullptr, 0, 2048, 128, 256, 2);   // sigmoid
  gate_combine<<<dim3(1024), dim3(256), 0, stream>>>(g_c, upd_c, h0_c, dfbuf, 0);
  // stream "next"
  gemm(xn_bf, 256, wn_t, 256, bn, h0_n, 128, cat_n, 256, 2048, 128, 256, 1);
  gemm(cat_n, 256, wmh_t, 128, bm, A_n, 128, nullptr, 0, 2048, 128, 128, 0);
  msg_maxagg<<<dim3(128, 16), dim3(128), 0, stream>>>(A_n, Bmsg, edge_w, cat_n);
  gemm(cat_n, 256, wu_t, 256, bu, upd_n, 128, nullptr, 0, 2048, 128, 256, 1);
  gemm(cat_n, 256, wg_t, 256, bg, g_n, 128, nullptr, 0, 2048, 128, 256, 2);
  gate_combine<<<dim3(1024), dim3(256), 0, stream>>>(g_n, upd_n, h0_n, dfbuf, 128);

  // --- classifier (decomposed uf@Wc1) ---
  gemm(e_bf, 32, wc1c_t, 32, bc1, Eb, 128, nullptr, 0, 16384, 128, 32, 0);        // Eb(+bc1)
  gemm(dfbuf + 128, 256, wc1a_t, 128, nullptr, Na, 128, nullptr, 0, 2048, 128, 128, 0); // nxt'@Wc1a
  gemm(dfbuf,       256, wc1b_t, 128, nullptr, Cb, 128, nullptr, 0, 2048, 128, 128, 0); // curr'@Wc1b
  logits_kernel<<<dim3(128, 16), dim3(128), 0, stream>>>(Na, Cb, Eb, Wc2, bc2, edge_w,
                                                         (float*)d_out);

  // --- dist predictor ---
  gemm(dfbuf, 256, wd1_t, 256, bd1, dh, 128, nullptr, 0, 2048, 128, 256, 1);
  dist_kernel<<<dim3(512), dim3(128), 0, stream>>>(dh, Wd2, bd2,
                                                   (float*)d_out + 16 * 128 * 128);
}
